// MinGRULayer_16784732193278
// MI455X (gfx1250) — matrix operand order repaired and checked by reading, not compiled, not run on hardware
//
#include <hip/hip_runtime.h>
#include <hip/hip_bf16.h>

// ---------------------------------------------------------------------------
// MinGRU layer for MI455X (gfx1250, wave32, WMMA + TDM)
//   B=4, T=4096, D=1024, Di=2048
//   GEMM1: hg[16384,4096] = x[16384,1024] @ W_hg[1024,4096]      (bf16 WMMA)
//   scan : h_t = sigmoid(-gate)*h_{t-1} + sigmoid(gate)*g(hidden) (f32)
//   GEMM2: out[16384,1024] = h[16384,2048] @ W_out[2048,1024]     (bf16 WMMA)
// Weights are converted to bf16 AND transposed once, so both GEMM operands
// stage as packed [128 rows][32 cols] 2D tiles -> native TDM tile loads.
// ---------------------------------------------------------------------------

typedef __bf16 v16bf __attribute__((ext_vector_type(16)));
typedef float  v8f   __attribute__((ext_vector_type(8)));
typedef unsigned int u32x4 __attribute__((ext_vector_type(4)));
typedef int          i32x4 __attribute__((ext_vector_type(4)));
typedef int          i32x8 __attribute__((ext_vector_type(8)));

#define BDIM  4
#define TDIM  4096
#define DDIM  1024
#define DI    2048
#define MROWS (BDIM * TDIM)   // 16384

// ---------------------------- f32 -> bf16 convert --------------------------
__global__ void cvt_bf16_kernel(const float* __restrict__ src,
                                __bf16* __restrict__ dst, int n) {
    int i = blockIdx.x * blockDim.x + threadIdx.x;
    if (i < n) dst[i] = (__bf16)src[i];
}

// ------------------- f32 -> bf16 convert with transpose --------------------
// dst[n][k] = (bf16) src[k][n];  src is [rows=K][cols=N] row-major.
__global__ __launch_bounds__(256)
void cvt_t_bf16_kernel(const float* __restrict__ src, __bf16* __restrict__ dst,
                       int rows, int cols) {
    __shared__ float tile[32][33];
    const int k0 = blockIdx.y * 32;
    const int n0 = blockIdx.x * 32;
    const int tx = threadIdx.x & 31;
    const int ty = threadIdx.x >> 5;     // 0..7
    #pragma unroll
    for (int i = ty; i < 32; i += 8)
        tile[i][tx] = src[(size_t)(k0 + i) * cols + n0 + tx];
    __syncthreads();
    #pragma unroll
    for (int i = ty; i < 32; i += 8)
        dst[(size_t)(n0 + i) * rows + k0 + tx] = (__bf16)tile[tx][i];
}

// ------------------------ TDM 2D tile load -> LDS ---------------------------
// Loads a [128 rows x 32 cols] tile of 2-byte elements starting at gaddr
// (row stride = row_stride_elems) into LDS at byte offset lds_off, packed.
// D# layout per CDNA5 ISA.  6-arg builtin form
// (clang-23 / therock-10.0 toolchain).
__device__ __forceinline__ void tdm_load_tile_128x32(unsigned lds_off,
                                                     const __bf16* gaddr,
                                                     unsigned row_stride_elems) {
    const unsigned long long ga = (unsigned long long)(uintptr_t)gaddr;
    u32x4 g0;
    g0[0] = 1u;                                            // count=1, user D#
    g0[1] = lds_off;                                       // lds_addr [63:32]
    g0[2] = (unsigned)(ga & 0xffffffffull);                // global_addr lo
    g0[3] = (unsigned)((ga >> 32) & 0x01ffffffull)         // global_addr hi
            | (2u << 30);                                  // type=2 ("image")
    const unsigned td0 = 32u;     // tensor_dim0 (elems)
    const unsigned td1 = 128u;    // tensor_dim1 (rows)
    const unsigned long long s0 = row_stride_elems;        // dim0 stride
    i32x8 g1;
    g1[0] = 0x00010000;                                    // data_size=1 (2B)
    g1[1] = (int)((td0 & 0xffffu) << 16);                  // tensor_dim0 lo
    g1[2] = (int)((td0 >> 16) & 0xffffu) | (int)((td1 & 0xffffu) << 16);
    g1[3] = (int)((td1 >> 16) & 0xffffu) | (int)(32u << 16);  // tile_dim0=32
    g1[4] = (int)128;                                      // tile_dim1=128
    g1[5] = (int)(s0 & 0xffffffffull);                     // dim0 stride lo
    g1[6] = (int)((s0 >> 32) & 0xffffull);                 // stride hi
    g1[7] = 0;
    i32x4 g2 = {0, 0, 0, 0};
    i32x4 g3 = {0, 0, 0, 0};
    i32x8 g4 = {0, 0, 0, 0, 0, 0, 0, 0};
    __builtin_amdgcn_tensor_load_to_lds(g0, g1, g2, g3, g4, 0);
}

// ------------------------- bf16 WMMA GEMM (f32 out) -------------------------
// C[M,N] = A[M,K] @ Wt[N,K]^T.  A row-major [M][K]; Wt row-major [N][K]
// (i.e. the pre-transposed B operand).  M%128==0, N%128==0, K%32==0.
// Block = 256 threads (8 wave32 waves). Block tile 128x128, K-step 32.
// Wave w computes rows [w*16, w*16+16) x 128 cols = 8 wmma 16x16 tiles.
// Tiles staged into double-buffered LDS by the Tensor Data Mover.
__global__ __launch_bounds__(256)
void gemm_bf16_wmma_kernel(const __bf16* __restrict__ A,
                           const __bf16* __restrict__ Wt,
                           float* __restrict__ C,
                           int M, int N, int K) {
    __shared__ alignas(16) __bf16 As[2][128][32];   // [m][k], packed
    __shared__ alignas(16) __bf16 Bs[2][128][32];   // [n][k], packed

    const int cbase = blockIdx.x * 128;
    const int rbase = blockIdx.y * 128;

    const int tid  = threadIdx.x;
    const int lane = tid & 31;        // wave32
    const int wid  = tid >> 5;        // 0..7
    const int half = lane >> 4;       // K-half selector (ISA 16-bit layouts)
    const int l15  = lane & 15;       // M (A frag) / N (B frag) index

    const __bf16* Ab = A  + (size_t)rbase * K;
    const __bf16* Bb = Wt + (size_t)cbase * K;

    const unsigned ldsA0 = (unsigned)(uintptr_t)&As[0][0][0];
    const unsigned ldsA1 = (unsigned)(uintptr_t)&As[1][0][0];
    const unsigned ldsB0 = (unsigned)(uintptr_t)&Bs[0][0][0];
    const unsigned ldsB1 = (unsigned)(uintptr_t)&Bs[1][0][0];

    v8f acc[8] = {};

    // prologue: DMA first K tile into buffer 0
    if (wid == 0) {
        tdm_load_tile_128x32(ldsA0, Ab, (unsigned)K);
        tdm_load_tile_128x32(ldsB0, Bb, (unsigned)K);
        __builtin_amdgcn_s_wait_tensorcnt(0);
    }
    __syncthreads();

    int cur = 0;
    for (int k0 = 0; k0 < K; k0 += 32) {
        // issue DMA for next K tile into the other buffer (overlaps compute)
        if (k0 + 32 < K && wid == 0) {
            tdm_load_tile_128x32(cur ? ldsA0 : ldsA1, Ab + k0 + 32, (unsigned)K);
            tdm_load_tile_128x32(cur ? ldsB0 : ldsB1, Bb + k0 + 32, (unsigned)K);
        }

        // ---- A fragment: 16x32 bf16, lane l15 = row, half selects K groups
        // element i -> K = half*16 + i (contiguous, matches B)
        v16bf afrag;
        {
            const __bf16* ar = &As[cur][wid * 16 + l15][0];
            #pragma unroll
            for (int i = 0; i < 8; ++i) afrag[i]     = ar[half * 16 + i];
            #pragma unroll
            for (int i = 0; i < 8; ++i) afrag[8 + i] = ar[half * 16 + 8 + i];
        }

        // ---- 8 column tiles: B fragment 32x16, lane l15 = col,
        //      element i -> K = half*16 + i (contiguous in Bs[n][k]) ----
        #pragma unroll
        for (int t = 0; t < 8; ++t) {
            v16bf bfrag;
            const __bf16* br = &Bs[cur][t * 16 + l15][half * 16];
            #pragma unroll
            for (int i = 0; i < 16; ++i) bfrag[i] = br[i];
            acc[t] = __builtin_amdgcn_wmma_f32_16x16x32_bf16(
                false, afrag, false, bfrag, (short)0, acc[t], false, false);
        }

        // publish next buffer: issuing wave waits for the DMA, then barrier
        if (wid == 0) __builtin_amdgcn_s_wait_tensorcnt(0);
        __syncthreads();
        cur ^= 1;
    }

    // ---- epilogue: C/D layout -> row = half*8 + r, col = l15 ----
    float* crow = C + (size_t)(rbase + wid * 16 + half * 8) * N + cbase + l15;
    #pragma unroll
    for (int t = 0; t < 8; ++t) {
        #pragma unroll
        for (int r = 0; r < 8; ++r) {
            crow[(size_t)r * N + t * 16] = acc[t][r];
        }
    }
}

// ----------------------- activation + sequential scan -----------------------
// One thread per (b, e) channel pair; loads across a warp are coalesced in e.
//   coeff_t = sigmoid(-gate) ; value_t = sigmoid(gate) * g(hidden)
//   g(x) = x + 0.5 (x>=0) else sigmoid(x)
__global__ __launch_bounds__(256)
void scan_kernel(const float* __restrict__ hg,
                 const float* __restrict__ prev_hidden,
                 __bf16* __restrict__ hbf,
                 float* __restrict__ next_hidden) {
    const int gid = blockIdx.x * blockDim.x + threadIdx.x; // 0..8191
    const int b = gid >> 11;         // / DI
    const int e = gid & (DI - 1);

    float h = fmaxf(prev_hidden[gid], 1e-8f);
    const float* base = hg + (size_t)b * TDIM * (2 * DI);

    for (int t = 0; t < TDIM; ++t) {
        const size_t ro = (size_t)t * (2 * DI);
        const float hl = base[ro + e];        // hidden pre-activation
        const float gg = base[ro + DI + e];   // gate pre-activation
        const float sig = 1.0f / (1.0f + __expf(-gg));   // sigmoid(gate)
        const float coeff = 1.0f - sig;                  // sigmoid(-gate)
        const float g = (hl >= 0.0f) ? (hl + 0.5f)
                                     : (1.0f / (1.0f + __expf(-hl)));
        h = coeff * h + sig * g;
        hbf[((size_t)b * TDIM + t) * DI + e] = (__bf16)h;
    }
    next_hidden[gid] = h;
}

// ------------------------------- launcher -----------------------------------
extern "C" void kernel_launch(void* const* d_in, const int* in_sizes, int n_in,
                              void* d_out, int out_size, void* d_ws, size_t ws_size,
                              hipStream_t stream) {
    const float* x    = (const float*)d_in[0];  // [4,4096,1024]
    const float* ph   = (const float*)d_in[1];  // [4,2048]
    const float* Whg  = (const float*)d_in[2];  // [1024,4096]
    const float* Wout = (const float*)d_in[3];  // [2048,1024]
    float* out = (float*)d_out;                 // [4,4096,1024] ++ [4,2048]

    // workspace layout (bytes)
    char* ws = (char*)d_ws;
    __bf16* xbf   = (__bf16*)(ws + 0);           // 16,777,216 * 2 =  32 MiB
    __bf16* whgt  = (__bf16*)(ws + 33554432);    // W_hg^T  [4096][1024] bf16
    __bf16* woutt = (__bf16*)(ws + 41943040);    // W_out^T [1024][2048] bf16
    float*  hg    = (float*) (ws + 46137344);    // 67,108,864 * 4 = 256 MiB
    __bf16* hbf   = (__bf16*)(ws + 314572800);   // 33,554,432 * 2 =  64 MiB

    // operand conversion (weights: convert + transpose to [N][K])
    cvt_bf16_kernel<<<MROWS * DDIM / 256, 256, 0, stream>>>(x, xbf, MROWS * DDIM);
    cvt_t_bf16_kernel<<<dim3(2 * DI / 32, DDIM / 32), 256, 0, stream>>>(
        Whg, whgt, DDIM, 2 * DI);
    cvt_t_bf16_kernel<<<dim3(DDIM / 32, DI / 32), 256, 0, stream>>>(
        Wout, woutt, DI, DDIM);

    // GEMM1: hg[16384,4096] = xbf[16384,1024] @ whgt[4096,1024]^T
    gemm_bf16_wmma_kernel<<<dim3(2 * DI / 128, MROWS / 128), 256, 0, stream>>>(
        xbf, whgt, hg, MROWS, 2 * DI, DDIM);

    // scan -> h (bf16) and next_hidden (f32, tail of d_out)
    scan_kernel<<<BDIM * DI / 256, 256, 0, stream>>>(
        hg, ph, hbf, out + (size_t)MROWS * DDIM);

    // GEMM2: out[16384,1024] = hbf[16384,2048] @ woutt[1024,2048]^T
    gemm_bf16_wmma_kernel<<<dim3(DDIM / 128, MROWS / 128), 256, 0, stream>>>(
        hbf, woutt, out, MROWS, DDIM, DI);
}
